// TreeLSTM_21603685499151
// MI455X (gfx1250) — compile-verified, matrix operand order checked by reading, not run on hardware
//
#include <hip/hip_runtime.h>
#include <hip/hip_bf16.h>

#define B_    512
#define S_    63
#define D_    256
#define VD_   512
#define R_    196
#define NOUT_ 3000
#define STK_  65      // S + 2 stack slots
#define NT    4       // n-tiles per wave (64 columns)

typedef __attribute__((ext_vector_type(16))) __bf16 bfrag;
typedef __attribute__((ext_vector_type(8)))  __bf16 bvec8;
typedef __attribute__((ext_vector_type(8)))  float  f32x8;
typedef __attribute__((ext_vector_type(4)))  float  f32x4;
typedef __attribute__((ext_vector_type(4)))  int    v4i;

#if defined(__has_builtin)
#if __has_builtin(__builtin_amdgcn_global_load_async_to_lds_b128) && \
    __has_builtin(__builtin_amdgcn_s_wait_asynccnt)
#define HAVE_ASYNC_LDS 1
#else
#define HAVE_ASYNC_LDS 0
#endif
#else
#define HAVE_ASYNC_LDS 0
#endif

// ---------------------------------------------------------------- WMMA core
static __device__ __forceinline__ f32x8 wmma_bf16(bfrag a, bfrag b, f32x8 c) {
  return __builtin_amdgcn_wmma_f32_16x16x32_bf16(false, a, false, b, (short)0, c,
                                                 false, false);
}

// 16x32 bf16 fragment from row-major [rows x ld] matrix (A layout; B symmetric).
// lane 0-15 : row r0+lane,    K = {k0..k0+7, k0+16..k0+23}
// lane 16-31: row r0+lane-16, K = {k0+8..k0+15, k0+24..k0+31}
static __device__ __forceinline__ bfrag load_frag(const __bf16* base, int ld,
                                                  int r0, int k0) {
  const int lane = threadIdx.x & 31;
  const int off  = (lane >> 4) << 3;
  const __bf16* p = base + (size_t)(r0 + (lane & 15)) * ld + k0 + off;
  bvec8 lo = *(const bvec8*)p;
  bvec8 hi = *(const bvec8*)(p + 16);
  bfrag f;
#pragma unroll
  for (int i = 0; i < 8; ++i) { f[i] = lo[i]; f[8 + i] = hi[i]; }
  return f;
}

// fragment from an LDS panel laid out [64 rows][32 k] bf16
static __device__ __forceinline__ bfrag frag_lds(const __bf16* Lb, int row, int off) {
  const __bf16* p = Lb + row * 32 + off;
  bvec8 lo = *(const bvec8*)p;
  bvec8 hi = *(const bvec8*)(p + 16);
  bfrag f;
#pragma unroll
  for (int i = 0; i < 8; ++i) { f[i] = lo[i]; f[8 + i] = hi[i]; }
  return f;
}

// A fragment from an f32 row (row pointer already lane-specific), cvt to bf16.
static __device__ __forceinline__ bfrag frag_f32_one(const float* row, int k0, int off) {
  const float* p = row + k0 + off;
  f32x4 a0 = *(const f32x4*)p;
  f32x4 a1 = *(const f32x4*)(p + 4);
  f32x4 b0 = *(const f32x4*)(p + 16);
  f32x4 b1 = *(const f32x4*)(p + 20);
  bfrag f;
#pragma unroll
  for (int i = 0; i < 4; ++i) {
    f[i]      = (__bf16)a0[i];
    f[4 + i]  = (__bf16)a1[i];
    f[8 + i]  = (__bf16)b0[i];
    f[12 + i] = (__bf16)b1[i];
  }
  return f;
}

static __device__ __forceinline__ bfrag frag_f32_sum(const float* r1, const float* r2,
                                                     int k0, int off) {
  const float* p1 = r1 + k0 + off;
  const float* p2 = r2 + k0 + off;
  f32x4 a0 = *(const f32x4*)p1;        f32x4 c0 = *(const f32x4*)p2;
  f32x4 a1 = *(const f32x4*)(p1 + 4);  f32x4 c1 = *(const f32x4*)(p2 + 4);
  f32x4 b0 = *(const f32x4*)(p1 + 16); f32x4 d0 = *(const f32x4*)(p2 + 16);
  f32x4 b1 = *(const f32x4*)(p1 + 20); f32x4 d1 = *(const f32x4*)(p2 + 20);
  bfrag f;
#pragma unroll
  for (int i = 0; i < 4; ++i) {
    f[i]      = (__bf16)(a0[i] + c0[i]);
    f[4 + i]  = (__bf16)(a1[i] + c1[i]);
    f[8 + i]  = (__bf16)(b0[i] + d0[i]);
    f[12 + i] = (__bf16)(b1[i] + d1[i]);
  }
  return f;
}

static __device__ __forceinline__ float act_apply(float v, int act) {
  if (act == 1) return tanhf(v);
  if (act == 2) return 1.0f / (1.0f + __expf(-v));
  return v;
}

// async global(16B) -> LDS copy, one chunk per thread
static __device__ __forceinline__ void async_copy16(const __bf16* g, __bf16* l) {
#if HAVE_ASYNC_LDS
  typedef __attribute__((address_space(1))) v4i* gv4i_p;
  typedef __attribute__((address_space(3))) v4i* lv4i_p;
  __builtin_amdgcn_global_load_async_to_lds_b128((gv4i_p)(void*)g, (lv4i_p)l, 0, 0);
#else
  *(f32x4*)l = *(const f32x4*)g;
#endif
}
static __device__ __forceinline__ void async_wait(int n) {
#if HAVE_ASYNC_LDS
  if (n == 0) __builtin_amdgcn_s_wait_asynccnt(0);
  else        __builtin_amdgcn_s_wait_asynccnt(1);
#else
  (void)n;
#endif
}

// ----------------------------------------------------- generic GEMM (+bias+act)
// Out[m,n] = act( sum_k A[m,k]*W[n,k] + bias[n] ).  A:[MxK] bf16, W:[NxK] bf16.
// Block = 8 waves = 128 M-rows x 64 N-cols.  B panel (64 x 32k) is staged into
// LDS once per block via double-buffered async global->LDS copies; each wave
// owns one 16-row M tile and 4 N tiles (A fragment reused 4x).
__global__ void gemm_bf16_kernel(const __bf16* __restrict__ A,
                                 const __bf16* __restrict__ W,
                                 const float* __restrict__ bias,
                                 float* __restrict__ outF,
                                 __bf16* __restrict__ outB,
                                 int M, int N, int K, int nGroups, int act) {
  __shared__ __bf16 lds[2][64 * 32];
  const int wid  = threadIdx.x >> 5;
  const int lane = threadIdx.x & 31;
  const int lr   = lane & 15;
  const int off  = (lane >> 4) << 3;
  const int mb = blockIdx.x / nGroups;
  const int ng = blockIdx.x % nGroups;
  const int m0 = (mb * 8 + wid) << 4;
  const int n0 = ng << 6;
  // staging role: thread -> (row 0..63, 8-element chunk)
  const int srow   = threadIdx.x >> 2;
  const int schunk = (threadIdx.x & 3) << 3;
  int gn = n0 + srow; if (gn >= N) gn = N - 1;
  const __bf16* gsrc = W + (size_t)gn * K + schunk;
  const int ldoff = srow * 32 + schunk;

  f32x8 acc[NT];
#pragma unroll
  for (int j = 0; j < NT; ++j) acc[j] = (f32x8){};

  const int nk = K >> 5;
  async_copy16(gsrc, &lds[0][ldoff]);          // preload k-slab 0
  bfrag a = load_frag(A, K, m0, 0);            // prefetch first A fragment
  for (int i = 0; i < nk; ++i) {
    if (i + 1 < nk) async_copy16(gsrc + ((size_t)(i + 1) << 5),
                                 &lds[(i + 1) & 1][ldoff]);
    async_wait(i + 1 < nk ? 1 : 0);
    __syncthreads();
    bfrag a_cur = a;
    if (i + 1 < nk) a = load_frag(A, K, m0, (i + 1) << 5);
    const __bf16* Lb = &lds[i & 1][0];
#pragma unroll
    for (int j = 0; j < NT; ++j) {
      bfrag b = frag_lds(Lb, j * 16 + lr, off);
      acc[j] = wmma_bf16(a_cur, b, acc[j]);
    }
    __syncthreads();
  }
#pragma unroll
  for (int j = 0; j < NT; ++j) {
    const int n = n0 + j * 16 + lr;
    const float bb = (n < N) ? bias[n] : 0.0f;
#pragma unroll
    for (int i = 0; i < 8; ++i) {
      const int m = m0 + i + ((lane >> 4) << 3);
      float v = act_apply(acc[j][i] + bb, act);
      if (n < N) {
        if (outF) outF[(size_t)m * N + n] = v;
        if (outB) outB[(size_t)m * N + n] = (__bf16)v;
      }
    }
  }
}

// -------------------------------------------- recurrence: 5 gate GEMMs per step
// z=0: i=sig(hsum Ui), z=1: o=sig(hsum Uo), z=2: u=tanh(hsum Uu),
// z=3: f1=sig(h1 Uf), z=4: f2=sig(h2 Uf).  M=512 (batch), N=K=256.
// Wave = 16 batch rows x 64 cols (A fragment + f32->bf16 cvt reused 4x).
__global__ void tree_gemm_kernel(const float* __restrict__ stack_h,
                                 const int* __restrict__ ptrs,
                                 const __bf16* __restrict__ Ui,
                                 const __bf16* __restrict__ Uo,
                                 const __bf16* __restrict__ Uu,
                                 const __bf16* __restrict__ Uf,
                                 const float* __restrict__ Ui_b,
                                 const float* __restrict__ Uo_b,
                                 const float* __restrict__ Uu_b,
                                 const float* __restrict__ Uf_b,
                                 float* __restrict__ gates) {
  const int wid  = threadIdx.x >> 5;
  const int tile = blockIdx.x * 8 + wid;        // 0..127  (32 tm x 4 ng)
  const int z    = blockIdx.y;                  // gate id 0..4
  const int tm = tile >> 2, ng = tile & 3;
  const int m0 = tm << 4, n0 = ng << 6;
  const int lane = threadIdx.x & 31;
  const int lr = lane & 15;
  const int off = (lane >> 4) << 3;
  const int b = m0 + lr;                        // this lane's batch row
  const int ptr = ptrs[b];
  const int r_idx = ptr - 1 > 0 ? ptr - 1 : 0;
  const int l_idx = ptr - 2 > 0 ? ptr - 2 : 0;
  const float* row1 = stack_h + ((size_t)b * STK_ + l_idx) * D_;  // h1
  const float* row2 = stack_h + ((size_t)b * STK_ + r_idx) * D_;  // h2
  const __bf16* W   = (z == 0) ? Ui : (z == 1) ? Uo : (z == 2) ? Uu : Uf;
  const float* bias = (z == 0) ? Ui_b : (z == 1) ? Uo_b : (z == 2) ? Uu_b : Uf_b;
  f32x8 acc[NT];
#pragma unroll
  for (int j = 0; j < NT; ++j) acc[j] = (f32x8){};
  for (int k0 = 0; k0 < D_; k0 += 32) {
    bfrag a;
    if (z < 3)       a = frag_f32_sum(row1, row2, k0, off);
    else if (z == 3) a = frag_f32_one(row1, k0, off);
    else             a = frag_f32_one(row2, k0, off);
#pragma unroll
    for (int j = 0; j < NT; ++j) {
      bfrag bf = load_frag(W, D_, n0 + j * 16, k0);
      acc[j] = wmma_bf16(a, bf, acc[j]);
    }
  }
  float* out = gates + (size_t)z * (B_ * D_);
#pragma unroll
  for (int j = 0; j < NT; ++j) {
    const int n = n0 + j * 16 + lr;
    const float bb = bias[n];
#pragma unroll
    for (int i = 0; i < 8; ++i) {
      const int m = m0 + i + ((lane >> 4) << 3);
      float v = acc[j][i] + bb;
      v = (z == 2) ? tanhf(v) : 1.0f / (1.0f + __expf(-v));
      out[(size_t)m * D_ + n] = v;
    }
  }
}

// LSTM cell + stack write for one step.  block = batch b, thread = feature d.
__global__ void step_cell_kernel(const float* __restrict__ gates,
                                 float* __restrict__ stack_h,
                                 float* __restrict__ stack_c,
                                 const int* __restrict__ ptrs,
                                 const int* __restrict__ ops,
                                 const float* __restrict__ buff_h,
                                 const float* __restrict__ buff_c,
                                 int t) {
  const int b = blockIdx.x, d = threadIdx.x;
  const int ptr = ptrs[b];
  const int r_idx = ptr - 1 > 0 ? ptr - 1 : 0;
  const int l_idx = ptr - 2 > 0 ? ptr - 2 : 0;
  const size_t base = (size_t)b * STK_ * D_;
  const float c1 = stack_c[base + (size_t)l_idx * D_ + d];
  const float c2 = stack_c[base + (size_t)r_idx * D_ + d];
  const size_t gi = (size_t)b * D_ + d;
  const size_t G = (size_t)B_ * D_;
  const float i_g = gates[0 * G + gi];
  const float o_g = gates[1 * G + gi];
  const float u_g = gates[2 * G + gi];
  const float f1  = gates[3 * G + gi];
  const float f2  = gates[4 * G + gi];
  const float c_new = i_g * u_g + f1 * c1 + f2 * c2;
  const float h_new = o_g * tanhf(c_new);
  const int op = ops[(size_t)b * S_ + t];
  if (op == 0) {
    const size_t bi = ((size_t)b * S_ + t) * D_ + d;
    stack_h[base + (size_t)ptr * D_ + d] = buff_h[bi];
    stack_c[base + (size_t)ptr * D_ + d] = buff_c[bi];
  } else if (op == 1) {
    stack_h[base + (size_t)l_idx * D_ + d] = h_new;
    stack_c[base + (size_t)l_idx * D_ + d] = c_new;
  }
}

// ------------------------------------------ attention GEMM + score reduction
// hA = tanh(vv @ WV^T + wqq[b]); scores[m] += sum_n hA[m,n]*WP[n]
__global__ void attn_score_kernel(const __bf16* __restrict__ vvb,
                                  const __bf16* __restrict__ WV,
                                  const float* __restrict__ wqq,
                                  const float* __restrict__ WP,
                                  float* __restrict__ scores) {
  const int wid  = threadIdx.x >> 5;
  const int tile = blockIdx.x * 8 + wid;        // 6272 tm x 4 ng
  const int tm = tile >> 2, ng = tile & 3;
  const int m0 = tm << 4, n0 = ng << 6;
  const int lane = threadIdx.x & 31;
  const int lr = lane & 15;
  f32x8 acc[NT];
#pragma unroll
  for (int j = 0; j < NT; ++j) acc[j] = (f32x8){};
  for (int k0 = 0; k0 < D_; k0 += 32) {
    bfrag a = load_frag(vvb, D_, m0, k0);
#pragma unroll
    for (int j = 0; j < NT; ++j) {
      bfrag b = load_frag(WV, D_, n0 + j * 16, k0);
      acc[j] = wmma_bf16(a, b, acc[j]);
    }
  }
  float s[8];
#pragma unroll
  for (int i = 0; i < 8; ++i) s[i] = 0.0f;
#pragma unroll
  for (int j = 0; j < NT; ++j) {
    const int n = n0 + j * 16 + lr;
    const float wp = WP[n];
#pragma unroll
    for (int i = 0; i < 8; ++i) {
      const int m = m0 + i + ((lane >> 4) << 3);
      const int mb = m / R_;
      const float hA = tanhf(acc[j][i] + wqq[(size_t)mb * D_ + n]);
      s[i] += hA * wp;
    }
  }
#pragma unroll
  for (int i = 0; i < 8; ++i) {
    float v = s[i];
    v += __shfl_xor(v, 1, 32);
    v += __shfl_xor(v, 2, 32);
    v += __shfl_xor(v, 4, 32);
    v += __shfl_xor(v, 8, 32);
    s[i] = v;
  }
  if (lr == 0) {
#pragma unroll
    for (int i = 0; i < 8; ++i) {
      const int m = m0 + i + ((lane >> 4) << 3);
      atomicAdd(&scores[m], s[i]);
    }
  }
}

// ------------------------------------------------------------- small kernels
__global__ void cvt_bf16_kernel(const float* __restrict__ s, __bf16* __restrict__ d,
                                long long n) {
  long long i = (long long)blockIdx.x * blockDim.x + threadIdx.x;
  if (i < n) d[i] = (__bf16)s[i];
}

__global__ void zero_f32_kernel(float* __restrict__ p, long long n) {
  long long i = (long long)blockIdx.x * blockDim.x + threadIdx.x;
  if (i < n) p[i] = 0.0f;
}

__global__ void gather_q_kernel(const float* __restrict__ wembed,
                                const int* __restrict__ questions,
                                __bf16* __restrict__ q) {
  const int row = blockIdx.x;   // b*S + s
  const int d = threadIdx.x;
  const int w = questions[row];
  q[(size_t)row * D_ + d] = (__bf16)wembed[(size_t)w * D_ + d];
}

__global__ void build_ptr_kernel(const int* __restrict__ ops, int* __restrict__ ptab) {
  const int b = blockIdx.x * blockDim.x + threadIdx.x;
  if (b >= B_) return;
  int ptr = 0;
  for (int t = 0; t < S_; ++t) {
    ptab[t * B_ + b] = ptr;
    const int op = ops[(size_t)b * S_ + t];
    ptr += (op == 0) ? 1 : ((op == 1) ? -1 : 0);
  }
  ptab[S_ * B_ + b] = ptr;   // final ptr
}

__global__ void leaf_cell_kernel(const float* __restrict__ ig,
                                 const float* __restrict__ ug,
                                 const float* __restrict__ og,
                                 float* __restrict__ bh, float* __restrict__ bc,
                                 long long n) {
  long long i = (long long)blockIdx.x * blockDim.x + threadIdx.x;
  if (i < n) {
    const float c = ig[i] * ug[i];
    bc[i] = c;
    bh[i] = og[i] * tanhf(c);
  }
}

__global__ void extract_r_kernel(const float* __restrict__ stack_h,
                                 const int* __restrict__ ptab,
                                 float* __restrict__ rF, __bf16* __restrict__ rB) {
  const int b = blockIdx.x, d = threadIdx.x;
  const int ptr = ptab[S_ * B_ + b];
  const int idx = ptr - 1 > 0 ? ptr - 1 : 0;
  const float v = stack_h[((size_t)b * STK_ + idx) * D_ + d];
  rF[(size_t)b * D_ + d] = v;
  rB[(size_t)b * D_ + d] = (__bf16)v;
}

__global__ void softmax196_kernel(const float* __restrict__ scores,
                                  float* __restrict__ p) {
  __shared__ float red[256];
  const int b = blockIdx.x, t = threadIdx.x;
  const float v = (t < R_) ? scores[(size_t)b * R_ + t] : -3.4e38f;
  red[t] = v; __syncthreads();
  for (int s = 128; s > 0; s >>= 1) {
    if (t < s) red[t] = fmaxf(red[t], red[t + s]);
    __syncthreads();
  }
  const float mx = red[0]; __syncthreads();
  const float e = (t < R_) ? __expf(v - mx) : 0.0f;
  red[t] = e; __syncthreads();
  for (int s = 128; s > 0; s >>= 1) {
    if (t < s) red[t] += red[t + s];
    __syncthreads();
  }
  const float inv = 1.0f / red[0];
  if (t < R_) p[(size_t)b * R_ + t] = e * inv;
}

__global__ void att_kernel(const float* __restrict__ p, const float* __restrict__ vv,
                           const float* __restrict__ r, float* __restrict__ attF,
                           __bf16* __restrict__ attB) {
  const int b = blockIdx.x, d = threadIdx.x;
  float acc = r[(size_t)b * D_ + d];
  const float* vb = vv + (size_t)b * R_ * D_ + d;
  const float* pb = p + (size_t)b * R_;
  for (int j = 0; j < R_; ++j) acc += pb[j] * vb[(size_t)j * D_];
  attF[(size_t)b * D_ + d] = acc;
  attB[(size_t)b * D_ + d] = (__bf16)acc;
}

// ---------------------------------------------------------------- host driver
extern "C" void kernel_launch(void* const* d_in, const int* in_sizes, int n_in,
                              void* d_out, int out_size, void* d_ws, size_t ws_size,
                              hipStream_t stream) {
  (void)in_sizes; (void)n_in; (void)out_size; (void)ws_size;
  const float* v         = (const float*)d_in[0];
  const int*   questions = (const int*)d_in[1];
  const int*   ops       = (const int*)d_in[2];
  const float* wembed    = (const float*)d_in[4];
  const float* W_w  = (const float*)d_in[5];  const float* W_b  = (const float*)d_in[6];
  const float* Wi_w = (const float*)d_in[7];  const float* Wi_b = (const float*)d_in[8];
  const float* Wu_w = (const float*)d_in[9];  const float* Wu_b = (const float*)d_in[10];
  const float* Wo_w = (const float*)d_in[11]; const float* Wo_b = (const float*)d_in[12];
  const float* Ui_w = (const float*)d_in[13]; const float* Ui_b = (const float*)d_in[14];
  const float* Uf_w = (const float*)d_in[15]; const float* Uf_b = (const float*)d_in[16];
  const float* Uo_w = (const float*)d_in[17]; const float* Uo_b = (const float*)d_in[18];
  const float* Uu_w = (const float*)d_in[19]; const float* Uu_b = (const float*)d_in[20];
  const float* WQ_w = (const float*)d_in[21]; const float* WQ_b = (const float*)d_in[22];
  const float* WV_w = (const float*)d_in[23];
  const float* WP_w = (const float*)d_in[24];
  const float* FC1_w = (const float*)d_in[26]; const float* FC1_b = (const float*)d_in[27];
  const float* FC2_w = (const float*)d_in[28]; const float* FC2_b = (const float*)d_in[29];
  float* out = (float*)d_out;

  // workspace bump allocator (256B aligned)
  size_t off = 0;
  auto alloc = [&](size_t bytes) -> void* {
    void* p = (char*)d_ws + off;
    off += (bytes + 255) & ~(size_t)255;
    return p;
  };
  const size_t MBR = (size_t)B_ * R_;           // 100352
  const size_t MBS = (size_t)B_ * S_;           // 32256
  __bf16* v_bf    = (__bf16*)alloc(MBR * VD_ * 2);
  __bf16* q_bf    = (__bf16*)alloc(MBS * D_ * 2);
  __bf16* Ww_bf   = (__bf16*)alloc((size_t)D_ * VD_ * 2);
  __bf16* Wi_bf   = (__bf16*)alloc((size_t)D_ * D_ * 2);
  __bf16* Wu_bf   = (__bf16*)alloc((size_t)D_ * D_ * 2);
  __bf16* Wo_bf   = (__bf16*)alloc((size_t)D_ * D_ * 2);
  __bf16* Ui_bf   = (__bf16*)alloc((size_t)D_ * D_ * 2);
  __bf16* Uf_bf   = (__bf16*)alloc((size_t)D_ * D_ * 2);
  __bf16* Uo_bf   = (__bf16*)alloc((size_t)D_ * D_ * 2);
  __bf16* Uu_bf   = (__bf16*)alloc((size_t)D_ * D_ * 2);
  __bf16* WQ_bf   = (__bf16*)alloc((size_t)D_ * D_ * 2);
  __bf16* WV_bf   = (__bf16*)alloc((size_t)D_ * D_ * 2);
  __bf16* FC1_bf  = (__bf16*)alloc((size_t)D_ * D_ * 2);
  __bf16* FC2_bf  = (__bf16*)alloc((size_t)NOUT_ * D_ * 2);
  float*  vv_f    = (float*)alloc(MBR * D_ * 4);
  __bf16* vv_bf   = (__bf16*)alloc(MBR * D_ * 2);
  float*  ig      = (float*)alloc(MBS * D_ * 4);
  float*  ug      = (float*)alloc(MBS * D_ * 4);
  float*  og      = (float*)alloc(MBS * D_ * 4);
  float*  buff_h  = (float*)alloc(MBS * D_ * 4);
  float*  buff_c  = (float*)alloc(MBS * D_ * 4);
  float*  stack_h = (float*)alloc((size_t)B_ * STK_ * D_ * 4);
  float*  stack_c = (float*)alloc((size_t)B_ * STK_ * D_ * 4);
  int*    ptab    = (int*)alloc((size_t)(S_ + 1) * B_ * 4);
  float*  gates   = (float*)alloc((size_t)5 * B_ * D_ * 4);
  float*  r_f     = (float*)alloc((size_t)B_ * D_ * 4);
  __bf16* r_bf    = (__bf16*)alloc((size_t)B_ * D_ * 2);
  float*  wqq     = (float*)alloc((size_t)B_ * D_ * 4);
  float*  scores  = (float*)alloc(MBR * 4);
  float*  pmat    = (float*)alloc(MBR * 4);
  float*  att_f   = (float*)alloc((size_t)B_ * D_ * 4);
  __bf16* att_bf  = (__bf16*)alloc((size_t)B_ * D_ * 2);
  __bf16* fc1_bf  = (__bf16*)alloc((size_t)B_ * D_ * 2);

  auto cvt = [&](const float* s, __bf16* d, long long n) {
    cvt_bf16_kernel<<<(int)((n + 255) / 256), 256, 0, stream>>>(s, d, n);
  };
  auto zero = [&](float* p, long long n) {
    zero_f32_kernel<<<(int)((n + 255) / 256), 256, 0, stream>>>(p, n);
  };
  // gemm launcher: grid = (M/128) * ceil(N/64) blocks of 256 threads
  auto gemm = [&](const __bf16* A, const __bf16* W, const float* bias,
                  float* oF, __bf16* oB, int M, int N, int K, int act) {
    int nGroups = (N + 63) / 64;
    gemm_bf16_kernel<<<(M >> 7) * nGroups, 256, 0, stream>>>(A, W, bias, oF, oB,
                                                             M, N, K, nGroups, act);
  };

  // ---- prep: conversions, gather, ptr table, zeroing
  cvt(v, v_bf, (long long)MBR * VD_);
  cvt(W_w, Ww_bf, (long long)D_ * VD_);
  cvt(Wi_w, Wi_bf, (long long)D_ * D_);  cvt(Wu_w, Wu_bf, (long long)D_ * D_);
  cvt(Wo_w, Wo_bf, (long long)D_ * D_);  cvt(Ui_w, Ui_bf, (long long)D_ * D_);
  cvt(Uf_w, Uf_bf, (long long)D_ * D_);  cvt(Uo_w, Uo_bf, (long long)D_ * D_);
  cvt(Uu_w, Uu_bf, (long long)D_ * D_);  cvt(WQ_w, WQ_bf, (long long)D_ * D_);
  cvt(WV_w, WV_bf, (long long)D_ * D_);  cvt(FC1_w, FC1_bf, (long long)D_ * D_);
  cvt(FC2_w, FC2_bf, (long long)NOUT_ * D_);
  gather_q_kernel<<<(int)MBS, D_, 0, stream>>>(wembed, questions, q_bf);
  build_ptr_kernel<<<2, 256, 0, stream>>>(ops, ptab);
  zero(stack_h, (long long)B_ * STK_ * D_);
  zero(stack_c, (long long)B_ * STK_ * D_);
  zero(scores, (long long)MBR);

  // ---- visual projection: vv = tanh(v @ W_w^T + W_b)  [100352 x 256]
  gemm(v_bf, Ww_bf, W_b, vv_f, vv_bf, (int)MBR, D_, VD_, /*tanh*/1);
  // ---- leaf gates on q  [32256 x 256]
  gemm(q_bf, Wi_bf, Wi_b, ig, nullptr, (int)MBS, D_, D_, 2);
  gemm(q_bf, Wu_bf, Wu_b, ug, nullptr, (int)MBS, D_, D_, 1);
  gemm(q_bf, Wo_bf, Wo_b, og, nullptr, (int)MBS, D_, D_, 2);
  leaf_cell_kernel<<<(int)((MBS * D_ + 255) / 256), 256, 0, stream>>>(
      ig, ug, og, buff_h, buff_c, (long long)MBS * D_);

  // ---- shift-reduce TreeLSTM: 63 sequential steps
  for (int t = 0; t < S_; ++t) {
    tree_gemm_kernel<<<dim3(16, 5), 256, 0, stream>>>(
        stack_h, ptab + t * B_, Ui_bf, Uo_bf, Uu_bf, Uf_bf,
        Ui_b, Uo_b, Uu_b, Uf_b, gates);
    step_cell_kernel<<<B_, D_, 0, stream>>>(gates, stack_h, stack_c,
                                            ptab + t * B_, ops, buff_h, buff_c, t);
  }
  extract_r_kernel<<<B_, D_, 0, stream>>>(stack_h, ptab, r_f, r_bf);

  // ---- attention
  gemm(r_bf, WQ_bf, WQ_b, wqq, nullptr, B_, D_, D_, 0);
  {
    int tiles = (int)(MBR / 16) * NT;   // 6272 tm x 4 ng
    attn_score_kernel<<<(tiles + 7) / 8, 256, 0, stream>>>(vv_bf, WV_bf, wqq,
                                                           WP_w, scores);
  }
  softmax196_kernel<<<B_, 256, 0, stream>>>(scores, pmat);
  att_kernel<<<B_, D_, 0, stream>>>(pmat, vv_f, r_f, att_f, att_bf);

  // ---- classifier head
  gemm(att_bf, FC1_bf, FC1_b, nullptr, fc1_bf, B_, D_, D_, 1);
  gemm(fc1_bf, FC2_bf, FC2_b, out, nullptr, B_, NOUT_, D_, 0);
}
